// multihead_self_attention_83030307766432
// MI455X (gfx1250) — matrix operand order repaired and checked by reading, not compiled, not run on hardware
//
#include <hip/hip_runtime.h>

#define DMODEL 1024
#define NHEADS 16
#define DK     64
#define SEQ    2048
#define BATCH  4

typedef __bf16 bf16;
typedef __attribute__((ext_vector_type(16))) __bf16       bf16x16;
typedef __attribute__((ext_vector_type(8)))  float        f32x8;
typedef __attribute__((ext_vector_type(4)))  unsigned int u32x4;

union FragU { bf16x16 h; u32x4 q[2]; };

// A-fragment: 16x32 bf16 tile, row-major, leading dim ld (elements).
// Lane layout (CDNA5): lane m = lane%16, kh = lane/16; VGPR0-3 hold K=kh*8..kh*8+7,
// VGPR4-7 hold K=16+kh*8..16+kh*8+7 -> two contiguous 16B loads.
__device__ inline bf16x16 load_frag_a(const bf16* __restrict__ p, int ld) {
  int lane = threadIdx.x & 31;
  int m = lane & 15, kh = lane >> 4;
  const bf16* r = p + (size_t)m * ld + kh * 8;
  FragU f;
  f.q[0] = *(const u32x4*)(r);
  f.q[1] = *(const u32x4*)(r + 16);
  return f.h;
}

// B-fragment: 32x16 tile where B[kk][nn] = p[nn*ld + kk] (kk contiguous).
// Lane layout: lane n = lane%16, kh = lane/16; K = kh*8..kh*8+7, 16+kh*8..16+kh*8+7 -> two 16B loads.
__device__ inline bf16x16 load_frag_b(const bf16* __restrict__ p, int ld) {
  int lane = threadIdx.x & 31;
  int n = lane & 15, kh = lane >> 4;
  const bf16* r = p + (size_t)n * ld + kh * 8;
  FragU f;
  f.q[0] = *(const u32x4*)(r);
  f.q[1] = *(const u32x4*)(r + 16);
  return f.h;
}

__device__ inline f32x8 wmma_bf16(bf16x16 a, bf16x16 b, f32x8 c) {
  return __builtin_amdgcn_wmma_f32_16x16x32_bf16(false, a, false, b, (short)0, c,
                                                 false, false);
}

__global__ __launch_bounds__(256) void f32_to_bf16(const float* __restrict__ in,
                                                   bf16* __restrict__ out, int n) {
  int i = blockIdx.x * blockDim.x + threadIdx.x;
  if (i < n) out[i] = (bf16)in[i];
}

// y = X * W^T  (X: [M,K] bf16 row-major, W: [N,K] bf16 row-major).
// MODE 0: fp32 row-major [M,N] out, MODE 1: bf16 head-split [B,H,S,DK],
// MODE 2: bf16 head-split transposed [B,H,DK,S].
template <int MODE>
__global__ __launch_bounds__(256) void gemm_xwt(const bf16* __restrict__ X,
                                                const bf16* __restrict__ W,
                                                void* __restrict__ out,
                                                int M, int N, int K, float scale) {
  int wave = threadIdx.x >> 5;
  int lane = threadIdx.x & 31;
  int wm = wave >> 2;            // 0..1
  int wn = wave & 3;             // 0..3
  int m0 = blockIdx.x * 64 + wm * 32;
  int n0 = blockIdx.y * 128 + wn * 32;

  f32x8 c00 = {}, c01 = {}, c10 = {}, c11 = {};
  for (int k0 = 0; k0 < K; k0 += 32) {
    bf16x16 a0 = load_frag_a(X + (size_t)m0 * K + k0, K);
    bf16x16 a1 = load_frag_a(X + (size_t)(m0 + 16) * K + k0, K);
    bf16x16 b0 = load_frag_b(W + (size_t)n0 * K + k0, K);
    bf16x16 b1 = load_frag_b(W + (size_t)(n0 + 16) * K + k0, K);
    c00 = wmma_bf16(a0, b0, c00);
    c01 = wmma_bf16(a0, b1, c01);
    c10 = wmma_bf16(a1, b0, c10);
    c11 = wmma_bf16(a1, b1, c11);
  }

  int kh = lane >> 4, n = lane & 15;
  f32x8 cc[2][2] = {{c00, c01}, {c10, c11}};
#pragma unroll
  for (int i = 0; i < 2; ++i)
#pragma unroll
    for (int j = 0; j < 2; ++j)
#pragma unroll
      for (int v = 0; v < 8; ++v) {
        int row = m0 + i * 16 + v + 8 * kh;   // M index
        int col = n0 + j * 16 + n;            // N index
        float val = cc[i][j][v] * scale;
        if (MODE == 0) {
          ((float*)out)[(size_t)row * N + col] = val;
        } else {
          int b = row >> 11, s = row & (SEQ - 1);
          int h = col >> 6, d = col & (DK - 1);
          if (MODE == 1)
            ((bf16*)out)[((size_t)((b * NHEADS + h) * SEQ + s)) * DK + d] = (bf16)val;
          else
            ((bf16*)out)[((size_t)((b * NHEADS + h) * DK + d)) * SEQ + s] = (bf16)val;
        }
      }
}

// Flash attention: Q,K in [B,H,S,DK] bf16 (Q pre-scaled by 1/sqrt(dk)),
// V transposed [B,H,DK,S] bf16. O: [B*S, D] bf16. One wave = 16 query rows.
__global__ __launch_bounds__(128) void attn_fwd(const bf16* __restrict__ Q,
                                                const bf16* __restrict__ Kt,
                                                const bf16* __restrict__ Vt,
                                                bf16* __restrict__ O) {
  __shared__ bf16 pLds[4][16][32];   // per-wave P staging tile (1 KB each)
  int wave = threadIdx.x >> 5;
  int lane = threadIdx.x & 31;
  int bh = blockIdx.x >> 5;          // S/64 = 32 query tiles per (b,h)
  int qt = blockIdx.x & 31;
  int q0 = qt * 64 + wave * 16;
  int b = bh >> 4, h = bh & 15;

  const bf16* Qb = Q + (size_t)bh * SEQ * DK;
  const bf16* Kb = Kt + (size_t)bh * SEQ * DK;
  const bf16* Vb = Vt + (size_t)bh * DK * SEQ;

  bf16x16 aq0 = load_frag_a(Qb + (size_t)q0 * DK + 0, DK);
  bf16x16 aq1 = load_frag_a(Qb + (size_t)q0 * DK + 32, DK);

  float mrow[8], lrow[8];
  f32x8 acc[4] = {};
#pragma unroll
  for (int v = 0; v < 8; ++v) { mrow[v] = -__builtin_inff(); lrow[v] = 0.0f; }

  int kh = lane >> 4, nn = lane & 15;

  for (int kb = 0; kb <= q0 + 15; kb += 32) {     // causal loop bound
    // scores: 16x32 = two 16x16 C tiles, K-dim = DK = 2 WMMA steps each
    f32x8 sc0 = {}, sc1 = {};
    {
      bf16x16 bk;
      bk = load_frag_b(Kb + (size_t)kb * DK + 0, DK);        sc0 = wmma_bf16(aq0, bk, sc0);
      bk = load_frag_b(Kb + (size_t)kb * DK + 32, DK);       sc0 = wmma_bf16(aq1, bk, sc0);
      bk = load_frag_b(Kb + (size_t)(kb + 16) * DK + 0, DK); sc1 = wmma_bf16(aq0, bk, sc1);
      bk = load_frag_b(Kb + (size_t)(kb + 16) * DK + 32, DK);sc1 = wmma_bf16(aq1, bk, sc1);
    }
    // online softmax (row m = v + 8*kh lives in the 16 lanes of this half-wave)
#pragma unroll
    for (int v = 0; v < 8; ++v) {
      int qrow = q0 + v + 8 * kh;
      if (kb + nn > qrow)      sc0[v] = -__builtin_inff();
      if (kb + 16 + nn > qrow) sc1[v] = -__builtin_inff();
      float t = fmaxf(sc0[v], sc1[v]);
#pragma unroll
      for (int off = 8; off; off >>= 1) t = fmaxf(t, __shfl_xor(t, off, 32));
      float nm = fmaxf(mrow[v], t);
      float alpha = __expf(mrow[v] - nm);
      float p0 = __expf(sc0[v] - nm);
      float p1 = __expf(sc1[v] - nm);
      float rs = p0 + p1;
#pragma unroll
      for (int off = 8; off; off >>= 1) rs += __shfl_xor(rs, off, 32);
      lrow[v] = lrow[v] * alpha + rs;
      mrow[v] = nm;
#pragma unroll
      for (int j = 0; j < 4; ++j) acc[j][v] *= alpha;
      pLds[wave][v + 8 * kh][nn] = (bf16)p0;          // C-layout -> row-major P
      pLds[wave][v + 8 * kh][16 + nn] = (bf16)p1;
    }
    asm volatile("s_wait_dscnt 0" ::: "memory");      // wave-local LDS RAW fence
    bf16x16 ap = load_frag_a(&pLds[wave][0][0], 32);  // P as 16x32 A-fragment
#pragma unroll
    for (int j = 0; j < 4; ++j) {                     // O += P * V  (dk tiled by 16)
      bf16x16 bv = load_frag_b(Vb + (size_t)(j * 16) * SEQ + kb, SEQ);
      acc[j] = wmma_bf16(ap, bv, acc[j]);
    }
  }

#pragma unroll
  for (int v = 0; v < 8; ++v) {
    float inv = 1.0f / lrow[v];
    int srow = q0 + v + 8 * kh;
    size_t base = ((size_t)(b * SEQ + srow)) * DMODEL + (size_t)h * DK;
#pragma unroll
    for (int j = 0; j < 4; ++j)
      O[base + j * 16 + nn] = (bf16)(acc[j][v] * inv);
  }
}

extern "C" void kernel_launch(void* const* d_in, const int* in_sizes, int n_in,
                              void* d_out, int out_size, void* d_ws, size_t ws_size,
                              hipStream_t stream) {
  const float* x  = (const float*)d_in[0];
  const float* wq = (const float*)d_in[1];
  const float* wk = (const float*)d_in[2];
  const float* wv = (const float*)d_in[3];
  const float* wo = (const float*)d_in[4];

  const size_t nX = (size_t)BATCH * SEQ * DMODEL;   // 8 Mi elements
  const size_t nW = (size_t)DMODEL * DMODEL;        // 1 Mi elements

  char* ws = (char*)d_ws;
  bf16* Xb  = (bf16*)ws; ws += nX * 2;
  bf16* WqB = (bf16*)ws; ws += nW * 2;
  bf16* WkB = (bf16*)ws; ws += nW * 2;
  bf16* WvB = (bf16*)ws; ws += nW * 2;
  bf16* WoB = (bf16*)ws; ws += nW * 2;
  bf16* Qh  = (bf16*)ws; ws += nX * 2;   // [B,H,S,DK], pre-scaled by 1/sqrt(dk)
  bf16* Kh  = (bf16*)ws; ws += nX * 2;   // [B,H,S,DK]
  bf16* Vt  = (bf16*)ws; ws += nX * 2;   // [B,H,DK,S]
  bf16* Ob  = (bf16*)ws; ws += nX * 2;   // [B*S, D]

  f32_to_bf16<<<(int)((nX + 255) / 256), 256, 0, stream>>>(x, Xb, (int)nX);
  f32_to_bf16<<<(int)((nW + 255) / 256), 256, 0, stream>>>(wq, WqB, (int)nW);
  f32_to_bf16<<<(int)((nW + 255) / 256), 256, 0, stream>>>(wk, WkB, (int)nW);
  f32_to_bf16<<<(int)((nW + 255) / 256), 256, 0, stream>>>(wv, WvB, (int)nW);
  f32_to_bf16<<<(int)((nW + 255) / 256), 256, 0, stream>>>(wo, WoB, (int)nW);

  dim3 grid(128, 8), blk(256);  // M=8192 -> 128 x 64-row tiles, N=1024 -> 8 x 128-col
  gemm_xwt<1><<<grid, blk, 0, stream>>>(Xb, WqB, Qh, 8192, 1024, 1024, 0.125f);
  gemm_xwt<1><<<grid, blk, 0, stream>>>(Xb, WkB, Kh, 8192, 1024, 1024, 1.0f);
  gemm_xwt<2><<<grid, blk, 0, stream>>>(Xb, WvB, Vt, 8192, 1024, 1024, 1.0f);

  attn_fwd<<<BATCH * NHEADS * (SEQ / 64), 128, 0, stream>>>(Qh, Kh, Vt, Ob);

  gemm_xwt<0><<<grid, blk, 0, stream>>>(Ob, WoB, d_out, 8192, 1024, 1024, 1.0f);
}
